// GeometricModel_85280870629519
// MI455X (gfx1250) — matrix operand order repaired and checked by reading, not compiled, not run on hardware
//
#include <hip/hip_runtime.h>

// MI455X / gfx1250 hyperbolic 2-layer forward.
// fp32 operands split into (hi,lo) _Float16 pairs; GEMM via
// v_wmma_f32_16x16x32_f16 with 3 cross products (ah*bh + ah*bl + al*bh) per
// k-step -> ~2^-22 rel error (fp32-faithful) at f16 matrix-core rates.
// A matrices stored in packed WMMA-fragment layout (32 lanes x 32B contiguous
// per 16x32 fragment) so GEMM A-loads are fully coalesced b128s.
// Each block owns a 64-row x 1024-col panel (halves L2 weight re-reads vs 32),
// accumulates in 256 VGPRs/wave, then runs the exact hyperbolic epilogue
// (gyro-matvec scale, Mobius bias add, tangent tanh) through a 263KB LDS panel.
// Workspace (~277 MB): Xp/Hp hi+lo (4x64MB) + W splits (8MB) + norms + bias.

typedef _Float16 v16h __attribute__((ext_vector_type(16)));
typedef float    v8f  __attribute__((ext_vector_type(8)));

#define MINN 1e-15f
#define MAXN 0.99999f            /* (1 - 1e-5) / sqrt(c), c = 1 */
#define DIMK 1024
#define LDSS 1028                /* 8*1028 % 64 == 32 -> conflict-free half-wave stores */
#define MBLK 64

__device__ __forceinline__ float atanh_clip(float z) {
    z = fminf(fmaxf(z, -1.0f + 1e-7f), 1.0f - 1e-7f);
    return atanhf(z);
}

__device__ __forceinline__ float wred(float v) {
#pragma unroll
    for (int o = 16; o > 0; o >>= 1) v += __shfl_xor(v, o, 32);
    return v;
}

// 32B contiguous fragment-lane load (packed A).
__device__ __forceinline__ v16h ld32B(const _Float16* p) {
    v16h r;
    uint4* d = reinterpret_cast<uint4*>(&r);
    const uint4* s = reinterpret_cast<const uint4*>(p);
    d[0] = s[0];
    d[1] = s[1];
    return r;
}

// Packed-A addressing: fragment (mtile,kt) holds rows [mtile*16,+16) x K [kt*32,+32).
// Lane (r%16)+16*hs stores halves idx0..15 = K { hs*8..hs*8+7, 16+hs*8..+7 }.
// Element offset in halves:
__device__ __forceinline__ size_t apk_off(int mtile, int kt, int lane_slot) {
    return ((size_t)(mtile * 32 + kt) * 32 + lane_slot) * 16;
}

union Pack16 { _Float16 h[16]; uint4 q[2]; };

// ---------------- prep kernels ----------------

__global__ void f32_to_f16split(const float* __restrict__ src,
                                _Float16* __restrict__ hi,
                                _Float16* __restrict__ lo, int n) {
    int stride = gridDim.x * blockDim.x;
    for (int i = blockIdx.x * blockDim.x + threadIdx.x; i < n; i += stride) {
        float v = src[i];
        _Float16 h = (_Float16)v;
        hi[i] = h;
        lo[i] = (_Float16)(v - (float)h);
    }
}

// One 64-thread block per row: emit packed-A fragments + row norm^2.
__global__ void prep_x(const float* __restrict__ x,
                       _Float16* __restrict__ xh, _Float16* __restrict__ xl,
                       float* __restrict__ xn2) {
    __shared__ float red[64];
    const int row = blockIdx.x, t = threadIdx.x;
    const size_t base = (size_t)row * DIMK;
    const int kt = t >> 1, hs = t & 1;
    const int k1 = kt * 32 + hs * 8;

    float v[16];
    const float4* p0 = reinterpret_cast<const float4*>(x + base + k1);
    const float4* p1 = reinterpret_cast<const float4*>(x + base + k1 + 16);
#pragma unroll
    for (int q = 0; q < 4; ++q) { v[q]     = p0[0][q]; v[q + 4]  = p0[1][q]; }
#pragma unroll
    for (int q = 0; q < 4; ++q) { v[q + 8] = p1[0][q]; v[q + 12] = p1[1][q]; }

    float s = 0.f;
    Pack16 ph, pl;
#pragma unroll
    for (int q = 0; q < 16; ++q) {
        s += v[q] * v[q];
        _Float16 h = (_Float16)v[q];
        ph.h[q] = h;
        pl.h[q] = (_Float16)(v[q] - (float)h);
    }
    const size_t off = apk_off(row >> 4, kt, (row & 15) + 16 * hs);
    uint4* dh = reinterpret_cast<uint4*>(xh + off);
    uint4* dl = reinterpret_cast<uint4*>(xl + off);
    dh[0] = ph.q[0]; dh[1] = ph.q[1];
    dl[0] = pl.q[0]; dl[1] = pl.q[1];

    red[t] = s;
    __syncthreads();
    for (int o = 32; o > 0; o >>= 1) {
        if (t < o) red[t] += red[t + o];
        __syncthreads();
    }
    if (t == 0) xn2[row] = red[0];
}

// yb = proj(expmap0(b)); yb[1024] holds ||yb||^2
__global__ void prep_bias(const float* __restrict__ b, float* __restrict__ yb) {
    __shared__ float red[256];
    const int tid = threadIdx.x;
    float s = 0.f;
    for (int j = tid; j < DIMK; j += 256) { float v = b[j]; s += v * v; }
    red[tid] = s;
    __syncthreads();
    for (int o = 128; o > 0; o >>= 1) {
        if (tid < o) red[tid] += red[tid + o];
        __syncthreads();
    }
    float bn2 = red[0];
    float bn  = fmaxf(sqrtf(bn2), MINN);
    float e   = tanhf(bn) / bn;          // expmap0 scale
    float en  = e * sqrtf(bn2);          // resulting norm
    float p   = (fmaxf(en, MINN) > MAXN) ? MAXN / fmaxf(en, MINN) : 1.f;
    float f   = e * p;
    for (int j = tid; j < DIMK; j += 256) yb[j] = f * b[j];
    if (tid == 0) yb[DIMK] = f * f * bn2;
}

// ---------------- fused layer: GEMM + hyperbolic epilogue ----------------
// grid = N/64 blocks, 256 threads (8 waves). Each wave: 4 M-tiles x 8 N-tiles
// (64 rows x 128 cols), so each B fragment from L2 feeds 4 WMMAs per product.

__global__ __launch_bounds__(256) void hyp_layer(
    const _Float16* __restrict__ Ah, const _Float16* __restrict__ Al,
    const _Float16* __restrict__ Bh, const _Float16* __restrict__ Bl,
    const float* __restrict__ xn2, const float* __restrict__ yb,
    _Float16* __restrict__ Oh, _Float16* __restrict__ Ol,
    float* __restrict__ on2, float* __restrict__ Of, int mode) {
    __shared__ float Cs[MBLK * LDSS];
    __shared__ float ybs[DIMK + 8];

    const int tid  = threadIdx.x;
    const int lane = tid & 31, wave = tid >> 5;
    const int half = lane >> 4, lm = lane & 15;
    const int rowBase  = blockIdx.x * MBLK;
    const int mTile0   = rowBase >> 4;
    const int nColBase = wave * 128;

    for (int j = tid; j <= DIMK; j += 256) ybs[j] = yb[j];

    v8f acc[4][8];
    {
        v8f z = {0.f, 0.f, 0.f, 0.f, 0.f, 0.f, 0.f, 0.f};
#pragma unroll
        for (int mt = 0; mt < 4; ++mt)
#pragma unroll
            for (int nt = 0; nt < 8; ++nt) acc[mt][nt] = z;
    }

    for (int kt = 0; kt < 32; ++kt) {
        v16h a_h[4], a_l[4];
#pragma unroll
        for (int mt = 0; mt < 4; ++mt) {
            size_t off = apk_off(mTile0 + mt, kt, lane);
            a_h[mt] = ld32B(Ah + off);
            a_l[mt] = ld32B(Al + off);
        }
#pragma unroll
        for (int nt = 0; nt < 8; ++nt) {
            size_t boff = (size_t)(nColBase + nt * 16 + lm) * DIMK + kt * 32 + half * 8;
            v16h b_h, b_l;
            reinterpret_cast<uint4*>(&b_h)[0] = *reinterpret_cast<const uint4*>(Bh + boff);
            reinterpret_cast<uint4*>(&b_h)[1] = *reinterpret_cast<const uint4*>(Bh + boff + 16);
            reinterpret_cast<uint4*>(&b_l)[0] = *reinterpret_cast<const uint4*>(Bl + boff);
            reinterpret_cast<uint4*>(&b_l)[1] = *reinterpret_cast<const uint4*>(Bl + boff + 16);
#pragma unroll
            for (int mt = 0; mt < 4; ++mt) {
                acc[mt][nt] = __builtin_amdgcn_wmma_f32_16x16x32_f16(
                    false, a_h[mt], false, b_h, (short)0, acc[mt][nt], false, false);
                acc[mt][nt] = __builtin_amdgcn_wmma_f32_16x16x32_f16(
                    false, a_h[mt], false, b_l, (short)0, acc[mt][nt], false, false);
                acc[mt][nt] = __builtin_amdgcn_wmma_f32_16x16x32_f16(
                    false, a_l[mt], false, b_h, (short)0, acc[mt][nt], false, false);
            }
        }
    }

    // C tile layout: VGPR r -> (M = mt*16 + r + 8*half, N = nColBase + nt*16 + lm)
#pragma unroll
    for (int mt = 0; mt < 4; ++mt)
#pragma unroll
        for (int nt = 0; nt < 8; ++nt)
#pragma unroll
            for (int r = 0; r < 8; ++r)
                Cs[(mt * 16 + r + half * 8) * LDSS + nColBase + nt * 16 + lm] =
                    acc[mt][nt][r];
    __syncthreads();

    // ---- per-row epilogue: 8 rows per wave ----
    const float y2 = ybs[DIMK];
    for (int i = 0; i < 8; ++i) {
        const int rl  = wave * 8 + i;
        const int row = rowBase + rl;

        float mm2 = 0.f, mdy = 0.f;
        for (int j = lane; j < DIMK; j += 32) {
            float v = Cs[rl * LDSS + j];
            mm2 += v * v;
            mdy += v * ybs[j];
        }
        mm2 = wred(mm2);
        mdy = wred(mdy);

        // gyro_matvec scaling
        float xn      = fmaxf(sqrtf(xn2[row]), MINN);
        float mxn_raw = sqrtf(mm2);
        float mxn     = fmaxf(mxn_raw, MINN);
        float s       = tanhf(mxn / xn * atanh_clip(xn)) / mxn;
        if (mm2 == 0.f) s = 0.f;
        // proj(res)
        {
            float n = fmaxf(s * mxn_raw, MINN);
            if (n > MAXN) s *= MAXN / n;
        }
        // gyro_add(res, yb):  h = A*m + B*yb
        float x2  = s * s * mm2;
        float xy  = s * mdy;
        float k1  = 1.f + 2.f * xy + y2;
        float den = fmaxf(1.f + 2.f * xy + x2 * y2, MINN);
        float A   = k1 * s / den;
        float B   = (1.f - x2) / den;
        // proj(h)
        float g2    = A * A * mm2 + 2.f * A * B * mdy + B * B * y2;
        float gn    = fmaxf(sqrtf(g2), MINN);
        float gpost = gn;
        if (gn > MAXN) {
            float p = MAXN / gn;
            A *= p; B *= p; gpost = MAXN;
        }
        // tangent tanh: t = tanh(artanh(|h|)/|h| * h)
        float xa  = fmaxf(gpost, MINN);
        float lam = atanh_clip(xa) / xa;
        float tn2 = 0.f;
        for (int j = lane; j < DIMK; j += 32) {
            float g = A * Cs[rl * LDSS + j] + B * ybs[j];
            float t = tanhf(lam * g);
            Cs[rl * LDSS + j] = t;
            tn2 += t * t;
        }
        tn2 = wred(tn2);
        // expmap0(t) + proj
        float tn  = fmaxf(sqrtf(tn2), MINN);
        float e   = tanhf(tn) / tn;
        float on  = tanhf(tn);
        float gam = (on > MAXN) ? e * (MAXN / on) : e;

        __syncthreads();   // phase-2 t values are read cross-lane below

        if (mode == 0) {
            // write packed-A fragments for next layer: 2 chunks/lane/array
#pragma unroll
            for (int cc = 0; cc < 2; ++cc) {
                const int c  = lane + cc * 32;
                const int kt = c >> 1, hs = c & 1;
                const int k1 = kt * 32 + hs * 8;
                const float4* q0 = reinterpret_cast<const float4*>(&Cs[rl * LDSS + k1]);
                const float4* q1 = reinterpret_cast<const float4*>(&Cs[rl * LDSS + k1 + 16]);
                float v[16];
#pragma unroll
                for (int q = 0; q < 4; ++q) { v[q]     = q0[0][q]; v[q + 4]  = q0[1][q]; }
#pragma unroll
                for (int q = 0; q < 4; ++q) { v[q + 8] = q1[0][q]; v[q + 12] = q1[1][q]; }
                Pack16 ph, pl;
#pragma unroll
                for (int q = 0; q < 16; ++q) {
                    float o = gam * v[q];
                    _Float16 h = (_Float16)o;
                    ph.h[q] = h;
                    pl.h[q] = (_Float16)(o - (float)h);
                }
                const size_t off = apk_off(row >> 4, kt, (row & 15) + 16 * hs);
                uint4* dh = reinterpret_cast<uint4*>(Oh + off);
                uint4* dl = reinterpret_cast<uint4*>(Ol + off);
                dh[0] = ph.q[0]; dh[1] = ph.q[1];
                dl[0] = pl.q[0]; dl[1] = pl.q[1];
            }
            if (lane == 0) on2[row] = gam * gam * tn2;
        } else {
            const size_t base = (size_t)row * DIMK;
#pragma unroll
            for (int jj = 0; jj < 8; ++jj) {
                const int j = jj * 128 + lane * 4;
                const float4* q = reinterpret_cast<const float4*>(&Cs[rl * LDSS + j]);
                float4 o;
#pragma unroll
                for (int q4 = 0; q4 < 4; ++q4) o[q4] = gam * (*q)[q4];
                *reinterpret_cast<float4*>(Of + base + j) = o;
            }
        }
    }
}

// ---------------- host ----------------

extern "C" void kernel_launch(void* const* d_in, const int* in_sizes, int n_in,
                              void* d_out, int out_size, void* d_ws, size_t ws_size,
                              hipStream_t stream) {
    (void)in_sizes; (void)n_in; (void)out_size; (void)ws_size;
    const float* x  = (const float*)d_in[0];
    const float* W1 = (const float*)d_in[1];
    const float* b1 = (const float*)d_in[2];
    const float* W2 = (const float*)d_in[3];
    const float* b2 = (const float*)d_in[4];
    float* out = (float*)d_out;

    const size_t N = 32768, D = 1024;
    char* ws = (char*)d_ws;
    size_t o = 0;
    _Float16* Xh = (_Float16*)(ws + o); o += N * D * 2;   // packed-A layout
    _Float16* Xl = (_Float16*)(ws + o); o += N * D * 2;
    _Float16* Hh = (_Float16*)(ws + o); o += N * D * 2;   // packed-A layout
    _Float16* Hl = (_Float16*)(ws + o); o += N * D * 2;
    _Float16* W1h = (_Float16*)(ws + o); o += D * D * 2;  // row-major
    _Float16* W1l = (_Float16*)(ws + o); o += D * D * 2;
    _Float16* W2h = (_Float16*)(ws + o); o += D * D * 2;
    _Float16* W2l = (_Float16*)(ws + o); o += D * D * 2;
    float* xn2 = (float*)(ws + o); o += N * 4;
    float* hn2 = (float*)(ws + o); o += N * 4;
    float* yb1 = (float*)(ws + o); o += 8192;
    float* yb2 = (float*)(ws + o); o += 8192;

    f32_to_f16split<<<1024, 256, 0, stream>>>(W1, W1h, W1l, (int)(D * D));
    f32_to_f16split<<<1024, 256, 0, stream>>>(W2, W2h, W2l, (int)(D * D));
    prep_x<<<(unsigned)N, 64, 0, stream>>>(x, Xh, Xl, xn2);
    prep_bias<<<1, 256, 0, stream>>>(b1, yb1);
    prep_bias<<<1, 256, 0, stream>>>(b2, yb2);

    hyp_layer<<<(unsigned)(N / MBLK), 256, 0, stream>>>(
        Xh, Xl, W1h, W1l, xn2, yb1, Hh, Hl, hn2, nullptr, 0);
    hyp_layer<<<(unsigned)(N / MBLK), 256, 0, stream>>>(
        Hh, Hl, W2h, W2l, hn2, yb2, nullptr, nullptr, nullptr, out, 1);
}
